// ViT_LSTM_Attention_53566832116172
// MI455X (gfx1250) — compile-verified, matrix operand order checked by reading, not run on hardware
//
#include <hip/hip_runtime.h>
#include <hip/hip_bf16.h>

// ViT->LSTM attention caption decoder for MI455X (gfx1250, wave32, WMMA).
// Strategy: one-time bf16 conversion + WMMA-tile packing of big weights
// (W_ea, [W_ih;W_hh], W_out) into d_ws, then all GEMMs via
// v_wmma_f32_16x16x32_bf16 with f32 accumulation. W_out (32.8MB bf16) stays
// resident in the 192MB L2 across the 40 sequential steps.
//
// d_ws usage (~47.5 MB): att1(6.5MB) + small state + packed weights (40.5MB).

typedef __bf16 bf16_t;
typedef __attribute__((ext_vector_type(16))) __bf16 v16bf;
typedef __attribute__((ext_vector_type(8)))  __bf16 v8bf;
typedef __attribute__((ext_vector_type(8)))  float  v8f;

#define B_  32
#define P_  197
#define ENC_ 768
#define EMB_ 512
#define HID_ 512
#define ATT_ 256
#define VOCAB_ 32000
#define SLEN_ 41
#define T_  40
#define KCAT_ 1792   // EMB + ENC + HID

// ---------------------------------------------------------------------------
// Pack f32 weight matrix (K x N, row-major; rows >= ksplit come from W2) into
// bf16 WMMA B-tiles. Tile (kt,nt) occupies 512 contiguous bf16; within a tile,
// lane l owns column (l&15) with k-range {s..s+7, s+16..s+23}, s=(l>=16)*8,
// matching the 16-bit A/B fragment layout (ISA 7.12.2) for a single
// 32-byte load per lane per tile in the GEMM.
// ---------------------------------------------------------------------------
__global__ void pack_weight(const float* __restrict__ W1,
                            const float* __restrict__ W2, int ksplit,
                            bf16_t* __restrict__ out, int K, int N)
{
    long idx = (long)blockIdx.x * blockDim.x + threadIdx.x;
    long total = (long)K * N;
    if (idx >= total) return;
    int t    = (int)(idx >> 9);
    int rrem = (int)(idx & 511);
    int l = rrem >> 4;
    int e = rrem & 15;
    int KT = K >> 5;
    int nt = t / KT;
    int kt = t - nt * KT;
    int n = nt * 16 + (l & 15);
    int k = kt * 32 + ((l >> 4) << 3) + (e < 8 ? e : e + 8);
    float v = (k < ksplit) ? W1[(long)k * N + n] : W2[(long)(k - ksplit) * N + n];
    out[idx] = (bf16_t)v;
}

// ---------------------------------------------------------------------------
// Generic WMMA bf16 GEMM: C[M,N] = A[M,K] @ B[K,N] (+bias), f32 accumulate.
// A is f32 (converted in-register) or bf16. B is pre-packed tiles (above).
// Each wave computes a 32x32 block: 4 accumulators, A loaded twice and each
// B tile loaded once per k-step. M%32==0, N%32==0, K%32==0 (true for all uses).
// ---------------------------------------------------------------------------
template <bool A_BF16>
__global__ void wmma_gemm(const void* __restrict__ Avoid,
                          const bf16_t* __restrict__ Bp,
                          float* __restrict__ C, const float* __restrict__ bias,
                          int M, int N, int K, long lda, long ldc)
{
    const int lane = threadIdx.x & 31;
    const int wave = threadIdx.x >> 5;
    const int ngroups = N >> 5;            // pairs of 16-col tiles
    const int mgroups = M >> 5;            // pairs of 16-row tiles
    int work = blockIdx.x * 8 + wave;      // wave-uniform -> EXEC stays full
    if (work >= mgroups * ngroups) return;
    const int mg = work / ngroups;
    const int ng = work - mg * ngroups;

    const int r  = lane & 15;              // row within 16-row tile
    const int ks = (lane >> 4) << 3;       // k phase: 0 or 8
    const int KT = K >> 5;

    v8f acc00 = {}, acc01 = {}, acc10 = {}, acc11 = {};

    for (int kt = 0; kt < KT; ++kt) {
        v16bf a0, a1;
        if (A_BF16) {
            const bf16_t* A  = (const bf16_t*)Avoid;
            const bf16_t* p0 = A + (long)(mg * 32 + r) * lda + kt * 32 + ks;
            const bf16_t* p1 = p0 + 16 * lda;
            v8bf lo0 = *(const v8bf*)p0, hi0 = *(const v8bf*)(p0 + 16);
            v8bf lo1 = *(const v8bf*)p1, hi1 = *(const v8bf*)(p1 + 16);
#pragma unroll
            for (int e = 0; e < 8; ++e) {
                a0[e] = lo0[e]; a0[e + 8] = hi0[e];
                a1[e] = lo1[e]; a1[e + 8] = hi1[e];
            }
        } else {
            const float* A  = (const float*)Avoid;
            const float* p0 = A + (long)(mg * 32 + r) * lda + kt * 32 + ks;
            const float* p1 = p0 + 16 * lda;
#pragma unroll
            for (int e = 0; e < 8; ++e) {
                a0[e]     = (bf16_t)p0[e];
                a0[e + 8] = (bf16_t)p0[e + 16];
                a1[e]     = (bf16_t)p1[e];
                a1[e + 8] = (bf16_t)p1[e + 16];
            }
        }
        const bf16_t* bt = Bp + ((long)(ng * 2) * KT + kt) * 512 + lane * 16;
        v16bf b0 = *(const v16bf*)bt;
        v16bf b1 = *(const v16bf*)(bt + (long)KT * 512);
        acc00 = __builtin_amdgcn_wmma_f32_16x16x32_bf16(false, a0, false, b0, (short)0, acc00, false, false);
        acc01 = __builtin_amdgcn_wmma_f32_16x16x32_bf16(false, a0, false, b1, (short)0, acc01, false, false);
        acc10 = __builtin_amdgcn_wmma_f32_16x16x32_bf16(false, a1, false, b0, (short)0, acc10, false, false);
        acc11 = __builtin_amdgcn_wmma_f32_16x16x32_bf16(false, a1, false, b1, (short)0, acc11, false, false);
    }

    // C/D layout (ISA 7.12.2): VGPR q -> row q (lanes 0-15) / row q+8 (16-31),
    // col = lane&15.
    const int row0 = mg * 32 + ((lane >> 4) << 3);
    const int col0 = ng * 32 + (lane & 15);
    const int col1 = col0 + 16;
    const float bv0 = bias ? bias[col0] : 0.0f;
    const float bv1 = bias ? bias[col1] : 0.0f;
#pragma unroll
    for (int q = 0; q < 8; ++q) {
        C[(long)(row0 + q)      * ldc + col0] = acc00[q] + bv0;
        C[(long)(row0 + q)      * ldc + col1] = acc01[q] + bv1;
        C[(long)(row0 + 16 + q) * ldc + col0] = acc10[q] + bv0;
        C[(long)(row0 + 16 + q) * ldc + col1] = acc11[q] + bv1;
    }
}

// ---------------------------------------------------------------------------
// h0/c0 from mean encoder output. One block per batch row.
// ---------------------------------------------------------------------------
__global__ void init_state(const float* __restrict__ enc,
                           const float* __restrict__ W_inh, const float* __restrict__ b_inh,
                           const float* __restrict__ W_inc, const float* __restrict__ b_inc,
                           float* __restrict__ h, float* __restrict__ c,
                           bf16_t* __restrict__ h_bf)
{
    __shared__ float mean_s[ENC_];
    const int b = blockIdx.x;
    for (int d = threadIdx.x; d < ENC_; d += 256) {
        float s = 0.0f;
        const float* p = enc + (long)b * P_ * ENC_ + d;
        for (int pp = 0; pp < P_; ++pp) s += p[(long)pp * ENC_];
        mean_s[d] = s * (1.0f / (float)P_);
    }
    __syncthreads();
    for (int d = threadIdx.x; d < HID_; d += 256) {
        float sh = b_inh[d], sc = b_inc[d];
        for (int k = 0; k < ENC_; ++k) {
            float m = mean_s[k];
            sh += m * W_inh[(long)k * HID_ + d];
            sc += m * W_inc[(long)k * HID_ + d];
        }
        h[b * HID_ + d] = sh;
        c[b * HID_ + d] = sc;
        h_bf[b * HID_ + d] = (bf16_t)sh;
    }
}

// ---------------------------------------------------------------------------
// Per-step attention + gate + LSTM-input assembly. One block per batch row.
// Produces bf16 row [x_t(512) | gate*awe(768) | h(512)] for the fused
// gates GEMM against packed [W_ih ; W_hh].
// ---------------------------------------------------------------------------
__global__ void att_step(const float* __restrict__ enc,
                         const int*   __restrict__ captions,
                         const float* __restrict__ emb,
                         const float* __restrict__ att1,
                         const float* __restrict__ W_da, const float* __restrict__ b_da,
                         const float* __restrict__ w_fa, const float* __restrict__ b_fa,
                         const float* __restrict__ W_fb, const float* __restrict__ b_fb,
                         const float* __restrict__ h,
                         bf16_t* __restrict__ lstm_in, int t)
{
    __shared__ float h_s[HID_];
    __shared__ float att2_s[ATT_];
    __shared__ float e_s[200];
    __shared__ float red_s[8];
    const int b   = blockIdx.x;
    const int tid = threadIdx.x;

    for (int d = tid; d < HID_; d += 256) h_s[d] = h[b * HID_ + d];
    __syncthreads();

    // att2 = h @ W_da + b_da  (one output per thread, coalesced on W_da cols)
    {
        float s = b_da[tid];
        for (int k = 0; k < HID_; ++k) s += h_s[k] * W_da[(long)k * ATT_ + tid];
        att2_s[tid] = s;
    }
    __syncthreads();

    // e_p = relu(att1 + att2) . w_fa + b_fa
    if (tid < P_) {
        const float* a1 = att1 + ((long)b * P_ + tid) * ATT_;
        float s = 0.0f;
        for (int a = 0; a < ATT_; ++a) {
            float v = a1[a] + att2_s[a];
            s += (v > 0.0f ? v : 0.0f) * w_fa[a];
        }
        e_s[tid] = s + b_fa[0];
    }
    __syncthreads();

    // softmax over 197
    float v = (tid < P_) ? e_s[tid] : -3.4e38f;
#pragma unroll
    for (int o = 16; o > 0; o >>= 1) v = fmaxf(v, __shfl_xor(v, o));
    if ((tid & 31) == 0) red_s[tid >> 5] = v;
    __syncthreads();
    if (tid == 0) {
        float m = red_s[0];
        for (int i = 1; i < 8; ++i) m = fmaxf(m, red_s[i]);
        red_s[0] = m;
    }
    __syncthreads();
    const float mx = red_s[0];
    __syncthreads();
    float ex = (tid < P_) ? __expf(e_s[tid] - mx) : 0.0f;
    float sv = ex;
#pragma unroll
    for (int o = 16; o > 0; o >>= 1) sv += __shfl_xor(sv, o);
    if ((tid & 31) == 0) red_s[tid >> 5] = sv;
    __syncthreads();
    if (tid == 0) {
        float s = 0.0f;
        for (int i = 0; i < 8; ++i) s += red_s[i];
        red_s[0] = s;
    }
    __syncthreads();
    const float inv = 1.0f / red_s[0];
    if (tid < P_) e_s[tid] = ex * inv;      // alpha
    __syncthreads();

    bf16_t* li = lstm_in + (long)b * KCAT_;

    // x_t and h copies
    const int tok = captions[b * SLEN_ + t];
    for (int d = tid; d < EMB_; d += 256) {
        li[d]            = (bf16_t)emb[(long)tok * EMB_ + d];
        li[EMB_ + ENC_ + d] = (bf16_t)h_s[d];
    }

    // awe_d = sum_p alpha_p enc[b,p,d];  gate_d = sigmoid(h @ W_fb + b_fb)
    for (int d = tid; d < ENC_; d += 256) {
        float aw = 0.0f;
        const float* ep = enc + (long)b * P_ * ENC_ + d;
        for (int p = 0; p < P_; ++p) aw += e_s[p] * ep[(long)p * ENC_];
        float g = b_fb[d];
        for (int k = 0; k < HID_; ++k) g += h_s[k] * W_fb[(long)k * ENC_ + d];
        g = 1.0f / (1.0f + __expf(-g));
        li[EMB_ + d] = (bf16_t)(aw * g);
    }
}

// ---------------------------------------------------------------------------
// LSTM pointwise: gates (pre-bias) -> h,c update (PyTorch gate order i,f,g,o).
// ---------------------------------------------------------------------------
__global__ void lstm_point(const float* __restrict__ gates,
                           const float* __restrict__ b_ih, const float* __restrict__ b_hh,
                           float* __restrict__ h, float* __restrict__ c,
                           bf16_t* __restrict__ h_bf)
{
    int idx = blockIdx.x * blockDim.x + threadIdx.x;
    if (idx >= B_ * HID_) return;
    int b = idx >> 9;
    int j = idx & (HID_ - 1);
    const float* g = gates + (long)b * 4 * HID_;
    float gi = g[j]             + b_ih[j]             + b_hh[j];
    float gf = g[HID_ + j]      + b_ih[HID_ + j]      + b_hh[HID_ + j];
    float gg = g[2 * HID_ + j]  + b_ih[2 * HID_ + j]  + b_hh[2 * HID_ + j];
    float go = g[3 * HID_ + j]  + b_ih[3 * HID_ + j]  + b_hh[3 * HID_ + j];
    float si = 1.0f / (1.0f + __expf(-gi));
    float sf = 1.0f / (1.0f + __expf(-gf));
    float so = 1.0f / (1.0f + __expf(-go));
    float cn = sf * c[idx] + si * tanhf(gg);
    float hn = so * tanhf(cn);
    c[idx] = cn;
    h[idx] = hn;
    h_bf[idx] = (bf16_t)hn;
}

// ---------------------------------------------------------------------------
extern "C" void kernel_launch(void* const* d_in, const int* in_sizes, int n_in,
                              void* d_out, int out_size, void* d_ws, size_t ws_size,
                              hipStream_t stream)
{
    const float* enc      = (const float*)d_in[0];
    const int*   captions = (const int*)  d_in[1];
    const float* emb      = (const float*)d_in[2];
    const float* W_ea     = (const float*)d_in[3];
    const float* b_ea     = (const float*)d_in[4];
    const float* W_da     = (const float*)d_in[5];
    const float* b_da     = (const float*)d_in[6];
    const float* w_fa     = (const float*)d_in[7];
    const float* b_fa     = (const float*)d_in[8];
    const float* W_inh    = (const float*)d_in[9];
    const float* b_inh    = (const float*)d_in[10];
    const float* W_inc    = (const float*)d_in[11];
    const float* b_inc    = (const float*)d_in[12];
    const float* W_fb     = (const float*)d_in[13];
    const float* b_fb     = (const float*)d_in[14];
    const float* W_ih     = (const float*)d_in[15];
    const float* b_ih     = (const float*)d_in[16];
    const float* W_hh     = (const float*)d_in[17];
    const float* b_hh     = (const float*)d_in[18];
    const float* W_out    = (const float*)d_in[19];
    const float* b_out    = (const float*)d_in[20];
    float* out = (float*)d_out;

    // carve workspace (256B aligned sections)
    char* wsp = (char*)d_ws;
    auto carve = [&](size_t bytes) -> void* {
        void* p = (void*)wsp;
        wsp += (bytes + 255) & ~(size_t)255;
        return p;
    };
    float*  att1    = (float*) carve((size_t)B_ * P_ * ATT_ * 4);        // 6.45 MB
    bf16_t* lstm_in = (bf16_t*)carve((size_t)B_ * KCAT_ * 2);
    float*  gates   = (float*) carve((size_t)B_ * 4 * HID_ * 4);
    float*  h       = (float*) carve((size_t)B_ * HID_ * 4);
    float*  c       = (float*) carve((size_t)B_ * HID_ * 4);
    bf16_t* h_bf    = (bf16_t*)carve((size_t)B_ * HID_ * 2);
    bf16_t* Wea_p   = (bf16_t*)carve((size_t)ENC_ * ATT_ * 2);           // 0.39 MB
    bf16_t* Wcat_p  = (bf16_t*)carve((size_t)KCAT_ * 4 * HID_ * 2);      // 7.34 MB
    bf16_t* Wout_p  = (bf16_t*)carve((size_t)HID_ * VOCAB_ * 2);         // 32.8 MB

    // one-time weight conversion + WMMA-tile packing (bf16)
    {
        long n1 = (long)ENC_ * ATT_;
        pack_weight<<<(int)((n1 + 255) / 256), 256, 0, stream>>>(W_ea, W_ea, ENC_, Wea_p, ENC_, ATT_);
        long n2 = (long)KCAT_ * 4 * HID_;
        pack_weight<<<(int)((n2 + 255) / 256), 256, 0, stream>>>(W_ih, W_hh, EMB_ + ENC_, Wcat_p, KCAT_, 4 * HID_);
        long n3 = (long)HID_ * VOCAB_;
        pack_weight<<<(int)((n3 + 255) / 256), 256, 0, stream>>>(W_out, W_out, HID_, Wout_p, HID_, VOCAB_);
    }

    // h0/c0
    init_state<<<B_, 256, 0, stream>>>(enc, W_inh, b_inh, W_inc, b_inc, h, c, h_bf);

    // att1 = enc @ W_ea + b_ea   (M=6304, N=256, K=768) — step-invariant
    {
        int waves = (B_ * P_ / 32) * (ATT_ / 32);   // 197 * 8 = 1576
        wmma_gemm<false><<<(waves + 7) / 8, 256, 0, stream>>>(
            (const void*)enc, Wea_p, att1, b_ea, B_ * P_, ATT_, ENC_, ENC_, ATT_);
    }

    // 40 sequential decode steps
    for (int t = 0; t < T_; ++t) {
        att_step<<<B_, 256, 0, stream>>>(enc, captions, emb, att1, W_da, b_da,
                                         w_fa, b_fa, W_fb, b_fb, h, lstm_in, t);

        // gates = [x_t|awe|h] @ [W_ih;W_hh]   (M=32, N=2048, K=1792)
        {
            int waves = (B_ / 32) * (4 * HID_ / 32);   // 64
            wmma_gemm<true><<<(waves + 7) / 8, 256, 0, stream>>>(
                (const void*)lstm_in, Wcat_p, gates, nullptr,
                B_, 4 * HID_, KCAT_, KCAT_, 4 * HID_);
        }

        lstm_point<<<(B_ * HID_ + 255) / 256, 256, 0, stream>>>(gates, b_ih, b_hh, h, c, h_bf);

        // preds_t = h_new @ W_out + b_out  (M=32, N=32000, K=512) -> d_out[:, t, :]
        {
            int waves = (B_ / 32) * (VOCAB_ / 32);     // 1000
            wmma_gemm<true><<<(waves + 7) / 8, 256, 0, stream>>>(
                (const void*)h_bf, Wout_p, out + (long)t * VOCAB_, b_out,
                B_, VOCAB_, HID_, HID_, (long)T_ * VOCAB_);
        }
    }
    (void)in_sizes; (void)n_in; (void)out_size; (void)ws_size;
}
